// LSNNCell_49117245997795
// MI455X (gfx1250) — matrix operand order repaired and checked by reading, not compiled, not run on hardware
//
#include <hip/hip_runtime.h>
#include <cstdint>

typedef __attribute__((ext_vector_type(16))) __bf16 v16bf;
typedef __attribute__((ext_vector_type(8)))  __bf16 v8bf;
typedef __attribute__((ext_vector_type(4)))  __bf16 v4bf;
typedef __attribute__((ext_vector_type(8)))  float  v8f;

#define BATCH    8192
#define IN_F     512
#define OUT_F    1024
#define KTOT     (IN_F + OUT_F)          // 1536 unified K
#define BM       128
#define BN       128
#define BK       32
#define LDSW     40                      // 32 + 8 pad halfs -> 80B row stride (16B aligned)
#define KT_TOTAL (KTOT / BK)             // 48
#define KT_SPLIT (IN_F / BK)             // 16
#define WCAT_BYTES ((size_t)OUT_F * KTOT * 2)   // 3 MiB bf16 weight concat

// LSNN constants
#define DT_TAU_MEM   0.1f        // DT * TAU_MEM_INV
#define DT_TAU_SYN   0.2f        // DT * TAU_SYN_INV
#define DT_TAU_ADAPT 1.25e-6f    // DT * TAU_ADAPT_INV
#define ADAPT_JUMP   0.00225f    // TAU_ADAPT_INV * BETA
#define V_TH_C       1.0f

// ---- CDNA5 async global->LDS staging (ASYNCcnt path) ----
__device__ __forceinline__ void async_load_b128(uint32_t lds_addr, uint32_t voff,
                                                const void* base) {
    // GVS mode: mem = SADDR(64b) + VADDR(32b offset); VDST = per-lane LDS byte address.
    asm volatile("global_load_async_to_lds_b128 %0, %1, %2"
                 :: "v"(lds_addr), "v"(voff), "s"(base)
                 : "memory");
}

__device__ __forceinline__ void wait_asynccnt0() {
#if __has_builtin(__builtin_amdgcn_s_wait_asynccnt)
    __builtin_amdgcn_s_wait_asynccnt(0);
#else
    asm volatile("s_wait_asynccnt 0x0" ::: "memory");
#endif
}

// Pass 1: Wcat[o][k] = bf16( k<512 ? Win[o][k] : Wrec[o][k-512] ), row stride 1536.
__global__ __launch_bounds__(256)
void convert_weights_kernel(const float* __restrict__ win,
                            const float* __restrict__ wrec,
                            __bf16* __restrict__ wcat) {
    int t = blockIdx.x * 256 + threadIdx.x;
    int e = t * 4;                       // quad of elements; KTOT%4==0, IN_F%4==0 -> no straddle
    if (e >= OUT_F * KTOT) return;
    int o = e / KTOT;
    int k = e - o * KTOT;
    const float* src = (k < IN_F) ? (win + (size_t)o * IN_F + k)
                                  : (wrec + (size_t)o * OUT_F + (k - IN_F));
    float4 f = *(const float4*)src;
    v4bf p;
    p.x = (__bf16)f.x; p.y = (__bf16)f.y; p.z = (__bf16)f.z; p.w = (__bf16)f.w;
    *(v4bf*)(wcat + (size_t)o * KTOT + k) = p;
}

template <bool USE_ASYNC>
__global__ __launch_bounds__(256)
void lsnn_step_kernel(const float* __restrict__ spikes,
                      const float* __restrict__ zin,
                      const float* __restrict__ vin,
                      const float* __restrict__ iin,
                      const float* __restrict__ bin,
                      const float* __restrict__ win,
                      const float* __restrict__ wrec,
                      const __bf16* __restrict__ wcat,
                      float* __restrict__ out)
{
    __shared__ __bf16 ldsA[BM][LDSW];
    __shared__ __bf16 ldsB[BN][LDSW];

    const int tid  = threadIdx.x;
    const int lane = tid & 31;
    const int wave = tid >> 5;
    const int wm   = wave & 3;    // 4 waves along M (32 rows each)
    const int wn   = wave >> 2;   // 2 waves along N (64 cols each)
    const int m16  = lane & 15;
    const int kh   = lane >> 4;   // which K-half this lane owns

    const int blockRow = blockIdx.x * BM;
    const int blockCol = blockIdx.y * BN;

    v8f c[2][4];
    #pragma unroll
    for (int mi = 0; mi < 2; ++mi)
        #pragma unroll
        for (int ni = 0; ni < 4; ++ni)
            c[mi][ni] = (v8f){0.f, 0.f, 0.f, 0.f, 0.f, 0.f, 0.f, 0.f};

    for (int kt = 0; kt < KT_TOTAL; ++kt) {
        const int kcolU = kt * BK;        // unified K column (for Wcat)
        // A side: fused K over [spikes | z]
        const float* aSrc;
        int aStride, kcolA;
        if (kt < KT_SPLIT) { aSrc = spikes; aStride = IN_F;  kcolA = kcolU; }
        else               { aSrc = zin;    aStride = OUT_F; kcolA = kcolU - IN_F; }

        // Load A tile (128x32 fp32) into registers (can overlap prior compute).
        float4 fa[4];
        float4 fb[4];
        #pragma unroll
        for (int j = 0; j < 4; ++j) {
            int idx = tid + j * 256;   // 0..1023
            int row = idx >> 3;        // 0..127
            int kq  = idx & 7;         // quad of floats
            fa[j] = *(const float4*)(aSrc + (size_t)(blockRow + row) * aStride + kcolA + kq * 4);
            if (!USE_ASYNC) {
                const float* bSrc;
                int bStride, kcolB;
                if (kt < KT_SPLIT) { bSrc = win;  bStride = IN_F;  kcolB = kcolU; }
                else               { bSrc = wrec; bStride = OUT_F; kcolB = kcolU - IN_F; }
                fb[j] = *(const float4*)(bSrc + (size_t)(blockCol + row) * bStride + kcolB + kq * 4);
            }
        }

        __syncthreads();   // all waves done reading previous LDS tiles

        if (USE_ASYNC) {
            // Stage B tile (128 rows x 64B of bf16 Wcat) via async global->LDS b128.
            #pragma unroll
            for (int j = 0; j < 2; ++j) {
                int idx = tid + j * 256;  // 0..511
                int row = idx >> 2;       // 0..127
                int seg = idx & 3;        // 16B segment within the 64B row chunk
                uint32_t voff = (uint32_t)((((blockCol + row) * KTOT) + kcolU + seg * 8) * 2);
                uint32_t laddr = (uint32_t)(uintptr_t)&ldsB[row][seg * 8];
                async_load_b128(laddr, voff, (const void*)wcat);
            }
        }

        // Convert + store A tile (and B tile in fallback mode).
        #pragma unroll
        for (int j = 0; j < 4; ++j) {
            int idx = tid + j * 256;
            int row = idx >> 3;
            int kq  = idx & 7;
            v4bf pa;
            pa.x = (__bf16)fa[j].x; pa.y = (__bf16)fa[j].y;
            pa.z = (__bf16)fa[j].z; pa.w = (__bf16)fa[j].w;
            *(v4bf*)&ldsA[row][kq * 4] = pa;
            if (!USE_ASYNC) {
                v4bf pb;
                pb.x = (__bf16)fb[j].x; pb.y = (__bf16)fb[j].y;
                pb.z = (__bf16)fb[j].z; pb.w = (__bf16)fb[j].w;
                *(v4bf*)&ldsB[row][kq * 4] = pb;
            }
        }
        if (USE_ASYNC) wait_asynccnt0();   // our async LDS writes landed
        __syncthreads();                   // everyone's tiles visible

        // WMMA fragments per ISA 16-bit layouts.
        // A (16x32 MxK): lane m16 = row; lanes 0-15: K 0-7 & 16-23, lanes 16-31: K 8-15 & 24-31.
        v16bf afrag[2], bfrag[4];
        #pragma unroll
        for (int mi = 0; mi < 2; ++mi) {
            int r = wm * 32 + mi * 16 + m16;
            union { v16bf v; v8bf h[2]; } u;
            u.h[0] = *(const v8bf*)&ldsA[r][kh * 8];
            u.h[1] = *(const v8bf*)&ldsA[r][kh * 8 + 16];
            afrag[mi] = u.v;
        }
        // B (32x16 KxN): lane m16 = column; lanes 0-15: K 0-7 & 16-23, lanes 16-31: K 8-15 & 24-31.
        #pragma unroll
        for (int ni = 0; ni < 4; ++ni) {
            int r = wn * 64 + ni * 16 + m16;
            union { v16bf v; v8bf h[2]; } u;
            u.h[0] = *(const v8bf*)&ldsB[r][kh * 8];
            u.h[1] = *(const v8bf*)&ldsB[r][kh * 8 + 16];
            bfrag[ni] = u.v;
        }

        #pragma unroll
        for (int mi = 0; mi < 2; ++mi)
            #pragma unroll
            for (int ni = 0; ni < 4; ++ni)
                c[mi][ni] = __builtin_amdgcn_wmma_f32_16x16x32_bf16(
                    false, afrag[mi], false, bfrag[ni],
                    (short)0, c[mi][ni], false, false);
    }

    // Fused LSNN epilogue. C layout: VGPR r, lanes 0-15 -> M=r, lanes 16-31 -> M=r+8; N = lane%16.
    const size_t NB = (size_t)BATCH * OUT_F;
    #pragma unroll
    for (int mi = 0; mi < 2; ++mi) {
        const int tm = blockRow + wm * 32 + mi * 16;
        #pragma unroll
        for (int ni = 0; ni < 4; ++ni) {
            const int tn   = blockCol + wn * 64 + ni * 16;
            const int col  = tn + m16;
            const int rowb = tm + kh * 8;
            #pragma unroll
            for (int r = 0; r < 8; ++r) {
                const int row = rowb + r;
                const size_t idx = (size_t)row * OUT_F + col;
                const float vv = vin[idx];
                const float iv = iin[idx];
                const float bv = bin[idx];
                const float v_dec = vv + DT_TAU_MEM * (iv - vv);      // v_leak = 0
                const float i_dec = iv - DT_TAU_SYN * iv;
                const float b_dec = bv + DT_TAU_ADAPT * (V_TH_C - bv);
                const float zn = (v_dec - b_dec > 0.0f) ? 1.0f : 0.0f;
                out[idx]          = zn;                                // z_new
                out[NB + idx]     = (1.0f - zn) * v_dec;               // v_new (v_reset = 0)
                out[2 * NB + idx] = i_dec + c[mi][ni][r];              // i_new
                out[3 * NB + idx] = b_dec + zn * ADAPT_JUMP;           // b_new
            }
        }
    }
}

extern "C" void kernel_launch(void* const* d_in, const int* in_sizes, int n_in,
                              void* d_out, int out_size, void* d_ws, size_t ws_size,
                              hipStream_t stream) {
    const float* spikes = (const float*)d_in[0];
    const float* z      = (const float*)d_in[1];
    const float* v      = (const float*)d_in[2];
    const float* i      = (const float*)d_in[3];
    const float* b      = (const float*)d_in[4];
    const float* win    = (const float*)d_in[5];
    const float* wrec   = (const float*)d_in[6];
    float* out = (float*)d_out;

    dim3 grid(BATCH / BM, OUT_F / BN);   // 64 x 8

    if (ws_size >= WCAT_BYTES) {
        __bf16* wcat = (__bf16*)d_ws;
        convert_weights_kernel<<<(OUT_F * KTOT / 4 + 255) / 256, 256, 0, stream>>>(
            win, wrec, wcat);
        lsnn_step_kernel<true><<<grid, 256, 0, stream>>>(
            spikes, z, v, i, b, win, wrec, wcat, out);
    } else {
        lsnn_step_kernel<false><<<grid, 256, 0, stream>>>(
            spikes, z, v, i, b, win, wrec, (const __bf16*)nullptr, out);
    }
}
